// TransformerBlock1_9887014715459
// MI455X (gfx1250) — matrix operand order repaired and checked by reading, not compiled, not run on hardware
//
#include <hip/hip_runtime.h>
#include <hip/hip_bf16.h>
#include <math.h>

// ---------------------------------------------------------------------------
// CDNA5 / gfx1250 types
// ---------------------------------------------------------------------------
typedef __attribute__((ext_vector_type(16))) __bf16 v16bf;
typedef __attribute__((ext_vector_type(8)))  __bf16 v8bf;
typedef __attribute__((ext_vector_type(4)))  __bf16 v4bf;
typedef __attribute__((ext_vector_type(2)))  __bf16 v2bf;
typedef __attribute__((ext_vector_type(8)))  float  v8f;

#define DIMC 192
#define HH   256
#define WW   192
#define HWN  (HH * WW)          // 49152
#define HEADS 8
#define CH   (DIMC / HEADS)     // 24
#define HID  (DIMC * 4)         // 768
#define QKVC (3 * DIMC)         // 576

__device__ __forceinline__ __bf16 f2bf(float f) { return (__bf16)f; }

__device__ __forceinline__ float gelu_exact(float v) {
    return 0.5f * v * (1.0f + erff(v * 0.70710678118654752f));
}

// ---------------------------------------------------------------------------
// Channel LayerNorm: per (b, h, w) position, reduce over c (stride HWN)
// ---------------------------------------------------------------------------
__global__ void ln_chan_kernel(const float* __restrict__ x,
                               const float* __restrict__ w,
                               const float* __restrict__ b,
                               float* __restrict__ y,
                               int total_pos /* B*HWN */) {
    int t = blockIdx.x * blockDim.x + threadIdx.x;
    if (t >= total_pos) return;
    int bb  = t / HWN;
    int pos = t - bb * HWN;
    const float* xp = x + (long long)bb * DIMC * HWN + pos;
    float*       yp = y + (long long)bb * DIMC * HWN + pos;
    float mu = 0.f;
    for (int c = 0; c < DIMC; ++c) mu += xp[(long long)c * HWN];
    mu *= (1.0f / DIMC);
    float var = 0.f;
    for (int c = 0; c < DIMC; ++c) {
        float d = xp[(long long)c * HWN] - mu;
        var += d * d;
    }
    var *= (1.0f / DIMC);
    float rs = rsqrtf(var + 1e-5f);
    for (int c = 0; c < DIMC; ++c) {
        yp[(long long)c * HWN] = (xp[(long long)c * HWN] - mu) * rs * w[c] + b[c];
    }
}

// ---------------------------------------------------------------------------
// FAST WMMA GEMM: C[m,n] = sum_k A[m,k]*B[k,n] (+resid)
// Hard assumptions (hold for qkv / proj):
//   aSk==1, bSn==1, cSn==1, M%32==0, N%128==0, K%32==0, simple z batching.
// Tile: 32(M) x 128(N) x 32(K); 256 threads = 8 waves (2Mx4N), 2 subtiles/wave.
// Unconditional float4 staging, packed (k,k+1) bf16 LDS stores, no guards.
// ---------------------------------------------------------------------------
__global__ void gemm_wmma_fast_kernel(
    const float* __restrict__ A, const float* __restrict__ Bm,
    float* __restrict__ C, const float* __restrict__ resid,
    int K,
    long long aBatch, int aSm,
    long long bBatch, long long bSk,
    long long cBatch, long long cSm) {

    __shared__ __attribute__((aligned(32))) __bf16 As[32 * 32];   // [m][k]  2 KB
    __shared__ __attribute__((aligned(32))) __bf16 Bs[128 * 32];  // [n][k]  8 KB

    const int tid  = threadIdx.x;
    const int lane = tid & 31;
    const int wid  = tid >> 5;
    const int l16  = lane & 15;
    const int half = lane >> 4;

    const int z  = blockIdx.z;
    const int bm = blockIdx.x * 32;
    const int bn = blockIdx.y * 128;
    const int wm = (wid >> 2) << 4;  // 0 or 16
    const int wn = (wid & 3) << 5;   // 0,32,64,96

    // staging coordinates
    const int am   = tid >> 3;         // 0..31     (A row within tile)
    const int ak4  = (tid & 7) << 2;   // 0,4,..28  (A k within tile, float4)
    const int bkk2 = (tid >> 5) << 1;  // 0,2,..14  (B even k row; +16 per i)
    const int bn4  = (tid & 31) << 2;  // 0..124    (B n within tile, float4)

    const float* ap = A + z * aBatch + (long long)(bm + am) * aSm + ak4;
    const float* bp = Bm + z * bBatch + (long long)bkk2 * bSk + bn + bn4;
    const long long offC = z * cBatch;

    v8f acc0 = {}, acc1 = {};

    for (int k0 = 0; k0 < K; k0 += 32) {
        // ---- unconditional vector staging ----
        float4 av = *(const float4*)ap;
        float4 bv[2][2];
#pragma unroll
        for (int i = 0; i < 2; ++i) {
            bv[i][0] = *(const float4*)(bp + (long long)(i * 16) * bSk);
            bv[i][1] = *(const float4*)(bp + (long long)(i * 16 + 1) * bSk);
        }
        ap += 32;
        bp += 32 * bSk;
        if (k0 + 32 < K) {
            __builtin_prefetch(ap, 0, 3);
            __builtin_prefetch(bp, 0, 3);
        }

        v4bf a4;
        a4[0] = f2bf(av.x); a4[1] = f2bf(av.y);
        a4[2] = f2bf(av.z); a4[3] = f2bf(av.w);
        *(v4bf*)&As[am * 32 + ak4] = a4;

#pragma unroll
        for (int i = 0; i < 2; ++i) {
            int kk = bkk2 + i * 16;  // even
            const float* e0 = (const float*)&bv[i][0];
            const float* e1 = (const float*)&bv[i][1];
#pragma unroll
            for (int j = 0; j < 4; ++j) {
                v2bf p;
                p[0] = f2bf(e0[j]);   // (n, kk)
                p[1] = f2bf(e1[j]);   // (n, kk+1)
                *(v2bf*)&Bs[(bn4 + j) * 32 + kk] = p;
            }
        }
        __syncthreads();

        // ---- fragments (contiguous k = 16*half + s, same in A and B) ----
        union { v16bf v; v8bf h[2]; } ua;
        const int kb = half * 16;
        ua.h[0] = *(const v8bf*)&As[(wm + l16) * 32 + kb];
        ua.h[1] = *(const v8bf*)&As[(wm + l16) * 32 + kb + 8];
        v16bf b0 = *(const v16bf*)&Bs[(wn + l16) * 32 + half * 16];
        v16bf b1 = *(const v16bf*)&Bs[(wn + 16 + l16) * 32 + half * 16];

        acc0 = __builtin_amdgcn_wmma_f32_16x16x32_bf16(
            false, ua.v, false, b0, (short)0, acc0, false, false);
        acc1 = __builtin_amdgcn_wmma_f32_16x16x32_bf16(
            false, ua.v, false, b1, (short)0, acc1, false, false);
        __syncthreads();
    }

    // ---- store (f32 16x16 C: VGPR r -> row half*8+r, col lane%16) ----
#pragma unroll
    for (int r = 0; r < 8; ++r) {
        int gm = bm + wm + half * 8 + r;
        long long base = offC + (long long)gm * cSm + bn + wn + l16;
        float v0 = acc0[r];
        float v1 = acc1[r];
        if (resid) {
            v0 += resid[base];
            v1 += resid[base + 16];
        }
        C[base]      = v0;
        C[base + 16] = v1;
    }
}

// ---------------------------------------------------------------------------
// GUARDED generic strided batched WMMA GEMM (for the tiny attention GEMMs)
// Per-z offsets: off = (z/zdiv)*outer + (z%zdiv)*inner + cnst
// Tile: 32x64x32, 8 waves, one 16x16 tile/wave.
// ---------------------------------------------------------------------------
__global__ void gemm_wmma_guarded_kernel(
    const float* __restrict__ A, const float* __restrict__ Bm,
    float* __restrict__ C,
    int M, int N, int K, int zdiv,
    long long aOuter, long long aInner, long long aCnst, int aSm, int aSk,
    long long bOuter, long long bInner, long long bCnst, long long bSk, long long bSn,
    long long cOuter, long long cInner, long long cCnst, long long cSm, long long cSn) {

    __shared__ __attribute__((aligned(32))) __bf16 As[32 * 32];
    __shared__ __attribute__((aligned(32))) __bf16 Bs[64 * 32];

    const int tid  = threadIdx.x;
    const int lane = tid & 31;
    const int wid  = tid >> 5;
    const int l16  = lane & 15;
    const int half = lane >> 4;

    const int z = blockIdx.z;
    const long long offA = (long long)(z / zdiv) * aOuter + (long long)(z % zdiv) * aInner + aCnst;
    const long long offB = (long long)(z / zdiv) * bOuter + (long long)(z % zdiv) * bInner + bCnst;
    const long long offC = (long long)(z / zdiv) * cOuter + (long long)(z % zdiv) * cInner + cCnst;

    const int bm = blockIdx.x * 32;
    const int bn = blockIdx.y * 64;
    const int wm = (wid >> 2) << 4;
    const int wn = (wid & 3) << 4;

    v8f acc = {};

    for (int k0 = 0; k0 < K; k0 += 32) {
        float va[4];
#pragma unroll
        for (int i = 0; i < 4; ++i) {
            int e = i * 256 + tid;
            int gm = bm + (e >> 5), gk = k0 + (e & 31);
            va[i] = (gm < M && gk < K)
                        ? A[offA + (long long)gm * aSm + (long long)gk * aSk]
                        : 0.0f;
        }
        float vb[8];
#pragma unroll
        for (int i = 0; i < 8; ++i) {
            int e = i * 256 + tid;
            int gk = k0 + (e >> 6), gn = bn + (e & 63);
            vb[i] = (gk < K && gn < N)
                        ? Bm[offB + (long long)gk * bSk + (long long)gn * bSn]
                        : 0.0f;
        }
#pragma unroll
        for (int i = 0; i < 4; ++i) {
            int e = i * 256 + tid;
            As[(e >> 5) * 32 + (e & 31)] = f2bf(va[i]);
        }
#pragma unroll
        for (int i = 0; i < 8; ++i) {
            int e = i * 256 + tid;
            Bs[(e & 63) * 32 + (e >> 6)] = f2bf(vb[i]);
        }
        __syncthreads();

        union { v16bf v; v8bf h[2]; } ua;
        const int kb = half * 16;
        ua.h[0] = *(const v8bf*)&As[(wm + l16) * 32 + kb];
        ua.h[1] = *(const v8bf*)&As[(wm + l16) * 32 + kb + 8];
        v16bf bfrag = *(const v16bf*)&Bs[(wn + l16) * 32 + half * 16];

        acc = __builtin_amdgcn_wmma_f32_16x16x32_bf16(
            false, ua.v, false, bfrag, (short)0, acc, false, false);
        __syncthreads();
    }

#pragma unroll
    for (int r = 0; r < 8; ++r) {
        int gm = bm + wm + half * 8 + r;
        int gn = bn + wn + l16;
        if (gm < M && gn < N) {
            C[offC + (long long)gm * cSm + (long long)gn * cSn] = acc[r];
        }
    }
}

// ---------------------------------------------------------------------------
// 3x3 depthwise conv, SAME zero padding, groups = QKVC
// ---------------------------------------------------------------------------
__global__ void dwconv_kernel(const float* __restrict__ in,
                              const float* __restrict__ w,
                              float* __restrict__ out, int total) {
    int idx = blockIdx.x * 256 + threadIdx.x;
    if (idx >= total) return;
    int x_ = idx % WW;
    int t1 = idx / WW;
    int y_ = t1 % HH;
    int t2 = t1 / HH;           // b*QKVC + ch
    int ch = t2 % QKVC;
    const float* base = in + (long long)t2 * HWN;
    const float* wc = w + ch * 9;
    float s = 0.f;
#pragma unroll
    for (int ky = 0; ky < 3; ++ky) {
        int yy = y_ + ky - 1;
        if (yy < 0 || yy >= HH) continue;
#pragma unroll
        for (int kx = 0; kx < 3; ++kx) {
            int xx = x_ + kx - 1;
            if (xx < 0 || xx >= WW) continue;
            s += base[yy * WW + xx] * wc[ky * 3 + kx];
        }
    }
    out[idx] = s;
}

// ---------------------------------------------------------------------------
// L2 norms of q (ch 0..191) and k (ch 192..383) rows of qkv_dw, per (b, ch')
// ---------------------------------------------------------------------------
__global__ void l2norm_kernel(const float* __restrict__ qkv2,
                              float* __restrict__ norms) {
    int cp = blockIdx.x;  // 0..383
    int b  = blockIdx.y;
    const float* p = qkv2 + ((long long)b * QKVC + cp) * HWN;
    float s = 0.f;
    for (int i = threadIdx.x; i < HWN; i += 256) {
        float v = p[i];
        s += v * v;
    }
    __shared__ float red[256];
    red[threadIdx.x] = s;
    __syncthreads();
    for (int st = 128; st > 0; st >>= 1) {
        if (threadIdx.x < st) red[threadIdx.x] += red[threadIdx.x + st];
        __syncthreads();
    }
    if (threadIdx.x == 0) norms[b * 384 + cp] = fmaxf(sqrtf(red[0]), 1e-12f);
}

// ---------------------------------------------------------------------------
// Softmax on 24x24 channel-attention logits, folding temperature and q/k norms
// ---------------------------------------------------------------------------
__global__ void softmax_kernel(float* __restrict__ attn,
                               const float* __restrict__ norms,
                               const float* __restrict__ temp) {
    int z = blockIdx.x;  // b*HEADS + head
    int b = z >> 3, hd = z & 7;
    int t = threadIdx.x;
    if (t < CH) {
        float tv = temp[hd];
        float qn = norms[b * 384 + hd * CH + t];
        float* row = attn + (long long)z * (CH * CH) + t * CH;
        float vals[CH];
        float mx = -1e30f;
        for (int d = 0; d < CH; ++d) {
            float kn = norms[b * 384 + DIMC + hd * CH + d];
            float v = row[d] * tv / (qn * kn);
            vals[d] = v;
            mx = fmaxf(mx, v);
        }
        float s = 0.f;
        for (int d = 0; d < CH; ++d) {
            vals[d] = expf(vals[d] - mx);
            s += vals[d];
        }
        float inv = 1.0f / s;
        for (int d = 0; d < CH; ++d) row[d] = vals[d] * inv;
    }
}

// ---------------------------------------------------------------------------
// Fused MLP over the LAST axis (w == DIMC): per 16 rows,
//   out = x1 + fc2( gelu( fc1(y) + b1 ) ) + b2, hidden kept in LDS as bf16.
// ---------------------------------------------------------------------------
__global__ void mlp_fused_kernel(const float* __restrict__ Y,
                                 const float* __restrict__ W1,
                                 const float* __restrict__ B1,
                                 const float* __restrict__ W2,
                                 const float* __restrict__ B2v,
                                 const float* __restrict__ X1,
                                 float* __restrict__ Out) {
    __shared__ __attribute__((aligned(32))) __bf16 Ys[16 * DIMC];  // 6 KB
    __shared__ __attribute__((aligned(32))) __bf16 Gs[16 * HID];   // 24 KB

    const int tid  = threadIdx.x;
    const int lane = tid & 31;
    const int wid  = tid >> 5;
    const int l16  = lane & 15;
    const int half = lane >> 4;
    const int kb   = half * 16;
    const long long row0 = (long long)blockIdx.x * 16;

#pragma unroll
    for (int i = 0; i < 12; ++i) {
        int e = i * 256 + tid;
        int m = e / DIMC, k = e - m * DIMC;
        Ys[m * DIMC + k] = f2bf(Y[(row0 + m) * DIMC + k]);
    }
    __syncthreads();

    // ---- phase 1: h1 = Y @ W1^T ; wave wid owns hidden cols [wid*96, wid*96+96) ----
    v8f acc[6];
#pragma unroll
    for (int t = 0; t < 6; ++t) acc[t] = (v8f){};

    for (int k0 = 0; k0 < DIMC; k0 += 32) {
        union { v16bf v; v8bf h[2]; } ua;
        ua.h[0] = *(const v8bf*)&Ys[l16 * DIMC + k0 + kb];
        ua.h[1] = *(const v8bf*)&Ys[l16 * DIMC + k0 + kb + 8];
#pragma unroll
        for (int t = 0; t < 6; ++t) {
            int n0 = (wid * 6 + t) * 16;
            const float* wp = W1 + (long long)(n0 + l16) * DIMC + k0 + half * 16;
            v16bf bf;
#pragma unroll
            for (int e2 = 0; e2 < 16; ++e2) bf[e2] = f2bf(wp[e2]);
            acc[t] = __builtin_amdgcn_wmma_f32_16x16x32_bf16(
                false, ua.v, false, bf, (short)0, acc[t], false, false);
        }
    }

#pragma unroll
    for (int t = 0; t < 6; ++t) {
        int n0 = (wid * 6 + t) * 16;
        int col = n0 + l16;
        float bcol = B1[col];
#pragma unroll
        for (int r = 0; r < 8; ++r) {
            int rr = half * 8 + r;
            float v = acc[t][r] + bcol;
            Gs[rr * HID + col] = f2bf(gelu_exact(v));
        }
    }
    __syncthreads();

    // ---- phase 2: out = G @ W2^T + b2 + x1 ----
    for (int nt = wid; nt < 12; nt += 8) {
        int n0 = nt * 16;
        v8f acc2 = {};
        for (int k0 = 0; k0 < HID; k0 += 32) {
            union { v16bf v; v8bf h[2]; } ua;
            ua.h[0] = *(const v8bf*)&Gs[l16 * HID + k0 + kb];
            ua.h[1] = *(const v8bf*)&Gs[l16 * HID + k0 + kb + 8];
            const float* wp = W2 + (long long)(n0 + l16) * HID + k0 + half * 16;
            v16bf bf;
#pragma unroll
            for (int e2 = 0; e2 < 16; ++e2) bf[e2] = f2bf(wp[e2]);
            acc2 = __builtin_amdgcn_wmma_f32_16x16x32_bf16(
                false, ua.v, false, bf, (short)0, acc2, false, false);
        }
        float bcol = B2v[n0 + l16];
#pragma unroll
        for (int r = 0; r < 8; ++r) {
            long long oi = (row0 + half * 8 + r) * DIMC + n0 + l16;
            Out[oi] = acc2[r] + bcol + X1[oi];
        }
    }
}

// ---------------------------------------------------------------------------
// Host-side orchestration
// ---------------------------------------------------------------------------
extern "C" void kernel_launch(void* const* d_in, const int* in_sizes, int n_in,
                              void* d_out, int out_size, void* d_ws, size_t ws_size,
                              hipStream_t stream) {
    const float* x      = (const float*)d_in[0];
    const float* ln1_w  = (const float*)d_in[1];
    const float* ln1_b  = (const float*)d_in[2];
    const float* temp   = (const float*)d_in[3];
    const float* qkv_w  = (const float*)d_in[4];
    const float* dw_w   = (const float*)d_in[5];
    const float* proj_w = (const float*)d_in[6];
    const float* ln2_w  = (const float*)d_in[7];
    const float* ln2_b  = (const float*)d_in[8];
    const float* fc1_w  = (const float*)d_in[9];
    const float* fc1_b  = (const float*)d_in[10];
    const float* fc2_w  = (const float*)d_in[11];
    const float* fc2_b  = (const float*)d_in[12];

    const int Bn = in_sizes[0] / (DIMC * HH * WW);  // 4
    const long long actElems = (long long)DIMC * HWN;
    const long long qkvElems = (long long)QKVC * HWN;

    float* ws = (float*)d_ws;
    float* wsY    = ws;
    float* wsX1   = wsY   + (long long)Bn * actElems;
    float* wsQKV  = wsX1  + (long long)Bn * actElems;
    float* wsQKV2 = wsQKV + (long long)Bn * qkvElems;
    float* wsAttn = wsQKV2 + (long long)Bn * qkvElems;
    float* wsNorm = wsAttn + (long long)Bn * HEADS * CH * CH;
    float* wsCtx  = wsQKV;  // reuse (qkv dead after dwconv)

    const long long sAct  = actElems;
    const long long sQkvB = qkvElems;
    const long long sHead = (long long)CH * HWN;

    // 1) LN1
    ln_chan_kernel<<<(Bn * HWN) / 256, 256, 0, stream>>>(x, ln1_w, ln1_b, wsY, Bn * HWN);

    // 2) qkv = qkv_w (576x192) @ y (192 x HWN)   [fast path]
    {
        dim3 g(QKVC / 32, HWN / 128, Bn);
        gemm_wmma_fast_kernel<<<g, 256, 0, stream>>>(
            qkv_w, wsY, wsQKV, nullptr, DIMC,
            /*aBatch*/0, DIMC, /*bBatch*/sAct, (long long)HWN,
            /*cBatch*/sQkvB, (long long)HWN);
    }

    // 3) depthwise 3x3 conv
    {
        int total = (int)((long long)Bn * qkvElems);
        dwconv_kernel<<<(total + 255) / 256, 256, 0, stream>>>(wsQKV, dw_w, wsQKV2, total);
    }

    // 4) L2 norms
    {
        dim3 g(2 * DIMC, Bn);
        l2norm_kernel<<<g, 256, 0, stream>>>(wsQKV2, wsNorm);
    }

    // 5) attn_raw[z] = q (24 x HWN) @ k^T (HWN x 24)   [guarded]
    {
        dim3 g(1, 1, Bn * HEADS);
        gemm_wmma_guarded_kernel<<<g, 256, 0, stream>>>(
            wsQKV2, wsQKV2, wsAttn,
            CH, CH, HWN, HEADS,
            /*A q*/ sQkvB, sHead, 0, HWN, 1,
            /*B k^T*/ sQkvB, sHead, (long long)DIMC * HWN, 1, (long long)HWN,
            /*C*/ (long long)HEADS * CH * CH, (long long)CH * CH, 0, CH, 1);
    }

    // 6) softmax with temperature / (||q|| ||k||)
    softmax_kernel<<<Bn * HEADS, 32, 0, stream>>>(wsAttn, wsNorm, temp);

    // 7) ctx[z] = attn (24x24) @ v (24 x HWN)   [guarded]
    {
        dim3 g(1, HWN / 64, Bn * HEADS);
        gemm_wmma_guarded_kernel<<<g, 256, 0, stream>>>(
            wsAttn, wsQKV2, wsCtx,
            CH, HWN, CH, HEADS,
            /*A*/ (long long)HEADS * CH * CH, (long long)CH * CH, 0, CH, 1,
            /*B v*/ sQkvB, sHead, (long long)2 * DIMC * HWN, (long long)HWN, 1,
            /*C*/ sAct, sHead, 0, (long long)HWN, 1);
    }

    // 8) x1 = x + proj_w (192x192) @ ctx   [fast path + residual]
    {
        dim3 g(DIMC / 32, HWN / 128, Bn);
        gemm_wmma_fast_kernel<<<g, 256, 0, stream>>>(
            proj_w, wsCtx, wsX1, x, DIMC,
            /*aBatch*/0, DIMC, /*bBatch*/sAct, (long long)HWN,
            /*cBatch*/sAct, (long long)HWN);
    }

    // 9) LN2
    ln_chan_kernel<<<(Bn * HWN) / 256, 256, 0, stream>>>(wsX1, ln2_w, ln2_b, wsY, Bn * HWN);

    // 10) fused MLP + residual -> d_out
    {
        int rows = Bn * DIMC * HH;  // 196608
        mlp_fused_kernel<<<rows / 16, 256, 0, stream>>>(
            wsY, fc1_w, fc1_b, fc2_w, fc2_b, wsX1, (float*)d_out);
    }
}
